// MultiHeadAttention_25520695673294
// MI455X (gfx1250) — matrix operand order repaired and checked by reading, not compiled, not run on hardware
//
#include <hip/hip_runtime.h>
#include <hip/hip_fp16.h>
#include <cstdint>

#define Bn 4
#define Tn 2048
#define Dn 1024
#define Hn 16
#define HDn 64

typedef __attribute__((ext_vector_type(16))) _Float16 v16h;
typedef __attribute__((ext_vector_type(8)))  _Float16 v8h;
typedef __attribute__((ext_vector_type(8)))  float    v8f;

static __device__ __forceinline__ v8f wmma_f16(v16h a, v16h b, v8f c) {
  // D = A(16x32 f16) * B(32x16 f16) + C(16x16 f32)
  return __builtin_amdgcn_wmma_f32_16x16x32_f16(false, a, false, b, (short)0, c,
                                                false, false);
}

// Truncated flat address of a __shared__ object == LDS byte offset
static __device__ __forceinline__ uint32_t lds_off(const void* p) {
  return (uint32_t)(uintptr_t)p;
}

// Async DMA of 16 bytes per lane: global -> LDS, tracked by ASYNCcnt.
static __device__ __forceinline__ void async_copy_b128(uint32_t lds, const void* g) {
  asm volatile("global_load_async_to_lds_b128 %0, %1, off"
               :: "v"(lds), "v"((uint64_t)(uintptr_t)g)
               : "memory");
}

static __device__ __forceinline__ void wait_async_le2() {
  asm volatile("s_wait_asynccnt 2" ::: "memory");
}
static __device__ __forceinline__ void wait_async_le0() {
  asm volatile("s_wait_asynccnt 0" ::: "memory");
}

// ---- Fragment loaders -------------------------------------------------------
// A-fragment (16x32, f16): lane<16 holds K {0-7,16-23}, lane>=16 holds {8-15,24-31}
// of row M = lane%16.  B-fragment (32x16, f16): lane<16 holds K {0-7,16-23}, lane>=16
// holds K {8-15,24-31}, of column N = lane%16.  All operands are K-contiguous in our
// memory layouts, so every lane does contiguous 16B loads (global or ds).

static __device__ __forceinline__ v16h afrag_f32(const float* __restrict__ rowk, int lane) {
  const int sel = (lane & 16) ? 8 : 0;
  const float4 x0 = *(const float4*)(rowk + sel);
  const float4 x1 = *(const float4*)(rowk + sel + 4);
  const float4 y0 = *(const float4*)(rowk + sel + 16);
  const float4 y1 = *(const float4*)(rowk + sel + 20);
  v16h r;
  r[0]=(_Float16)x0.x; r[1]=(_Float16)x0.y; r[2]=(_Float16)x0.z; r[3]=(_Float16)x0.w;
  r[4]=(_Float16)x1.x; r[5]=(_Float16)x1.y; r[6]=(_Float16)x1.z; r[7]=(_Float16)x1.w;
  r[8]=(_Float16)y0.x; r[9]=(_Float16)y0.y; r[10]=(_Float16)y0.z; r[11]=(_Float16)y0.w;
  r[12]=(_Float16)y1.x; r[13]=(_Float16)y1.y; r[14]=(_Float16)y1.z; r[15]=(_Float16)y1.w;
  return r;
}

static __device__ __forceinline__ v16h bfrag_f32(const float* __restrict__ rowk, int lane) {
  const float* p = rowk + ((lane & 16) ? 8 : 0);
  const float4 a = *(const float4*)(p);
  const float4 b = *(const float4*)(p + 4);
  const float4 c = *(const float4*)(p + 16);
  const float4 d = *(const float4*)(p + 20);
  v16h r;
  r[0]=(_Float16)a.x;  r[1]=(_Float16)a.y;  r[2]=(_Float16)a.z;  r[3]=(_Float16)a.w;
  r[4]=(_Float16)b.x;  r[5]=(_Float16)b.y;  r[6]=(_Float16)b.z;  r[7]=(_Float16)b.w;
  r[8]=(_Float16)c.x;  r[9]=(_Float16)c.y;  r[10]=(_Float16)c.z; r[11]=(_Float16)c.w;
  r[12]=(_Float16)d.x; r[13]=(_Float16)d.y; r[14]=(_Float16)d.z; r[15]=(_Float16)d.w;
  return r;
}

static __device__ __forceinline__ v16h afrag_f16(const _Float16* rowk, int lane) {
  const int sel = (lane & 16) ? 8 : 0;
  const v8h a = *(const v8h*)(rowk + sel);
  const v8h b = *(const v8h*)(rowk + sel + 16);
  v16h r;
#pragma unroll
  for (int i = 0; i < 8; ++i) { r[i] = a[i]; r[8 + i] = b[i]; }
  return r;
}

static __device__ __forceinline__ v16h bfrag_f16(const _Float16* rowk, int lane) {
  const _Float16* p = rowk + ((lane & 16) ? 8 : 0);
  const v8h a = *(const v8h*)(p);
  const v8h b = *(const v8h*)(p + 16);
  v16h r;
#pragma unroll
  for (int i = 0; i < 8; ++i) { r[i] = a[i]; r[8 + i] = b[i]; }
  return r;
}

// ---- QKV projection: C[M=8192,N=1024] = X @ W^T + bias, stored f16 ----------
// MODE 0/1: out layout [B,H,T,HD]   (Q and K, head-split)
// MODE 2  : out layout [B,H,HD,T]   (V transposed -> PV GEMM becomes NT)
template <int MODE>
__global__ __launch_bounds__(256) void proj_kernel(const float* __restrict__ X,
                                                   const float* __restrict__ W,
                                                   const float* __restrict__ bias,
                                                   _Float16* __restrict__ out) {
  const int lane = threadIdx.x & 31;
  const int wv   = threadIdx.x >> 5;
  const int row0 = (blockIdx.x * 8 + wv) * 16;   // 16-row M tile
  const int n0   = blockIdx.y * 64;              // 64-col N tile (4 WMMA tiles)
  const int li   = lane & 15;
  const int half8 = (lane & 16) ? 8 : 0;

  const float* arow = X + (size_t)(row0 + li) * Dn;
  v8f acc0 = {}, acc1 = {}, acc2 = {}, acc3 = {};

  for (int k0 = 0; k0 < Dn; k0 += 32) {
    v16h a = afrag_f32(arow + k0, lane);
    acc0 = wmma_f16(a, bfrag_f32(W + (size_t)(n0 +  0 + li) * Dn + k0, lane), acc0);
    acc1 = wmma_f16(a, bfrag_f32(W + (size_t)(n0 + 16 + li) * Dn + k0, lane), acc1);
    acc2 = wmma_f16(a, bfrag_f32(W + (size_t)(n0 + 32 + li) * Dn + k0, lane), acc2);
    acc3 = wmma_f16(a, bfrag_f32(W + (size_t)(n0 + 48 + li) * Dn + k0, lane), acc3);
  }

  v8f acc[4] = {acc0, acc1, acc2, acc3};
#pragma unroll
  for (int t = 0; t < 4; ++t) {
    const int col = n0 + t * 16 + li;
    const float bc = bias[col];
    const int h = col >> 6, hd = col & 63;
#pragma unroll
    for (int v = 0; v < 8; ++v) {
      const int row = row0 + v + half8;       // global M row = b*T + t_idx
      const int b = row >> 11, tt = row & 2047;
      const float val = acc[t][v] + bc;
      size_t addr;
      if (MODE == 2) addr = ((size_t)((b * Hn + h) * HDn + hd)) * Tn + tt;
      else           addr = ((size_t)((b * Hn + h) * Tn + tt)) * HDn + hd;
      out[addr] = (_Float16)val;
    }
  }
}

// ---- Flash attention --------------------------------------------------------
// One block = one (b,h) x 128 queries (8 waves x 16-query tiles).  K/V chunks
// (32 keys) are DMA'd into LDS once per block via GLOBAL_LOAD_ASYNC_TO_LDS_B128
// (double-buffered, ASYNCcnt), then all 8 waves feed WMMA fragments from LDS.
// Computes S^T = K*Q^T so exp(S^T) lands directly in the PV A-fragment layout.
__global__ __launch_bounds__(256) void attn_kernel(const _Float16* __restrict__ Qh,
                                                   const _Float16* __restrict__ Kh,
                                                   const _Float16* __restrict__ Vt,
                                                   _Float16* __restrict__ AO) {
  __shared__ _Float16 sK[2][32 * HDn];   // [j=32][d=64]  4KB per buffer
  __shared__ _Float16 sV[2][HDn * 32];   // [d=64][j=32]  4KB per buffer

  const int tid  = threadIdx.x;
  const int lane = tid & 31;
  const int wv   = tid >> 5;
  const int bh   = blockIdx.x;                       // 0..63
  const int i0   = (blockIdx.y * 8 + wv) * 16;       // query tile base
  const int li   = lane & 15;
  const int half8 = (lane & 16) ? 8 : 0;

  const _Float16* kbase = Kh + (size_t)bh * Tn * HDn;
  const _Float16* vbase = Vt + (size_t)bh * HDn * Tn;

  // Per-thread 16B DMA slices.  K chunk rows jc..jc+31 are contiguous 4KB in
  // [B,H,T,HD]; V chunk is 64 segments of 64B from [B,H,HD,T].
  const int kslice = tid * 8;                        // halves within 4KB chunk
  const int vseg = tid >> 2, vpart = (tid & 3) * 8;  // segment d, 16B part

  auto issue_chunk = [&](int jc, int buf) {
    async_copy_b128(lds_off(&sK[buf][kslice]),
                    kbase + (size_t)jc * HDn + kslice);
    async_copy_b128(lds_off(&sV[buf][vseg * 32 + vpart]),
                    vbase + (size_t)vseg * Tn + jc + vpart);
  };

  // Q as B-fragment (Q^T columns are Q rows, d-contiguous), two K=32 steps.
  const _Float16* qrow = Qh + ((size_t)bh * Tn + i0 + li) * HDn;
  const v16h qb0 = bfrag_f16(qrow, lane);        // d = 0..31
  const v16h qb1 = bfrag_f16(qrow + 32, lane);   // d = 32..63

  v8f o0 = {}, o1 = {}, o2 = {}, o3 = {};
  float m_r = -3.0e38f, s_r = 0.0f;
  const float sc = 0.125f;                       // 1/sqrt(64)

  issue_chunk(0, 0);

  for (int jc = 0, n = 0; jc < Tn; jc += 32, ++n) {
    const int buf = n & 1;
    if (jc + 32 < Tn) {
      issue_chunk(jc + 32, buf ^ 1);  // prefetch next chunk into other buffer
      wait_async_le2();               // my chunk-n slices landed
    } else {
      wait_async_le0();
    }
    __syncthreads();                  // all waves' chunk-n slices landed

    // S^T tiles from LDS: rows j (VGPR dim), cols i (lane dim)
    const _Float16* kc = sK[buf];
    const _Float16* vc = sV[buf];
    const _Float16* k0r = kc + li * HDn;
    const _Float16* k1r = kc + (16 + li) * HDn;
    v8f st0 = {}, st1 = {};
    st0 = wmma_f16(afrag_f16(k0r, lane),      qb0, st0);
    st0 = wmma_f16(afrag_f16(k0r + 32, lane), qb1, st0);
    st1 = wmma_f16(afrag_f16(k1r, lane),      qb0, st1);
    st1 = wmma_f16(afrag_f16(k1r + 32, lane), qb1, st1);

    // online softmax over rows i = lane%16 (stats replicated across halves)
    float mx = -3.0e38f;
#pragma unroll
    for (int v = 0; v < 8; ++v) mx = fmaxf(mx, fmaxf(st0[v], st1[v]));
    mx *= sc;
    mx = fmaxf(mx, __shfl_xor(mx, 16, 32));
    const float mn = fmaxf(m_r, mx);
    const float alpha = __expf(m_r - mn);
    float p0[8], p1[8], ls = 0.0f;
#pragma unroll
    for (int v = 0; v < 8; ++v) {
      p0[v] = __expf(st0[v] * sc - mn);
      p1[v] = __expf(st1[v] * sc - mn);
      ls += p0[v] + p1[v];
    }
    ls += __shfl_xor(ls, 16, 32);
    s_r = s_r * alpha + ls;
    m_r = mn;

    // P already in A-fragment packing (lane = query row, slots = key K values)
    v16h pa;
#pragma unroll
    for (int v = 0; v < 8; ++v) { pa[v] = (_Float16)p0[v]; pa[8 + v] = (_Float16)p1[v]; }

    // rescale O rows (rows live in VGPR dim -> bpermute alpha per row)
#pragma unroll
    for (int v = 0; v < 8; ++v) {
      const float av = __shfl(alpha, v + half8, 32);
      o0[v] *= av; o1[v] *= av; o2[v] *= av; o3[v] *= av;
    }

    // O += P @ V  (V rows in LDS are j-contiguous, stride 32 halves)
    o0 = wmma_f16(pa, bfrag_f16(vc + ( 0 + li) * 32, lane), o0);
    o1 = wmma_f16(pa, bfrag_f16(vc + (16 + li) * 32, lane), o1);
    o2 = wmma_f16(pa, bfrag_f16(vc + (32 + li) * 32, lane), o2);
    o3 = wmma_f16(pa, bfrag_f16(vc + (48 + li) * 32, lane), o3);

    __syncthreads();                  // done reading buf before it is reused
  }

  const float inv = 1.0f / s_r;
  float si[8];
#pragma unroll
  for (int v = 0; v < 8; ++v) si[v] = __shfl(inv, v + half8, 32);

  // merged-head store: AO[b, t, h*64 + d]  (f16, A-operand of output proj)
  const int bb = bh >> 4, h = bh & 15;
  v8f oacc[4] = {o0, o1, o2, o3};
#pragma unroll
  for (int t = 0; t < 4; ++t) {
#pragma unroll
    for (int v = 0; v < 8; ++v) {
      const int row = i0 + v + half8;
      const int d = t * 16 + li;
      AO[((size_t)(bb * Tn + row)) * Dn + h * HDn + d] = (_Float16)(oacc[t][v] * si[v]);
    }
  }
}

// ---- Output projection: out[M,N] = AO(f16) @ Wo^T + bo, f32 out -------------
__global__ __launch_bounds__(256) void oproj_kernel(const _Float16* __restrict__ A,
                                                    const float* __restrict__ W,
                                                    const float* __restrict__ bias,
                                                    float* __restrict__ out) {
  const int lane = threadIdx.x & 31;
  const int wv   = threadIdx.x >> 5;
  const int row0 = (blockIdx.x * 8 + wv) * 16;
  const int n0   = blockIdx.y * 64;
  const int li   = lane & 15;
  const int half8 = (lane & 16) ? 8 : 0;

  const _Float16* arow = A + (size_t)(row0 + li) * Dn;
  v8f acc0 = {}, acc1 = {}, acc2 = {}, acc3 = {};

  for (int k0 = 0; k0 < Dn; k0 += 32) {
    v16h a = afrag_f16(arow + k0, lane);
    acc0 = wmma_f16(a, bfrag_f32(W + (size_t)(n0 +  0 + li) * Dn + k0, lane), acc0);
    acc1 = wmma_f16(a, bfrag_f32(W + (size_t)(n0 + 16 + li) * Dn + k0, lane), acc1);
    acc2 = wmma_f16(a, bfrag_f32(W + (size_t)(n0 + 32 + li) * Dn + k0, lane), acc2);
    acc3 = wmma_f16(a, bfrag_f32(W + (size_t)(n0 + 48 + li) * Dn + k0, lane), acc3);
  }

  v8f acc[4] = {acc0, acc1, acc2, acc3};
#pragma unroll
  for (int t = 0; t < 4; ++t) {
    const int col = n0 + t * 16 + li;
    const float bc = bias[col];
#pragma unroll
    for (int v = 0; v < 8; ++v) {
      const int row = row0 + v + half8;
      out[(size_t)row * Dn + col] = acc[t][v] + bc;
    }
  }
}

extern "C" void kernel_launch(void* const* d_in, const int* in_sizes, int n_in,
                              void* d_out, int out_size, void* d_ws, size_t ws_size,
                              hipStream_t stream) {
  const float* q  = (const float*)d_in[0];
  const float* k  = (const float*)d_in[1];
  const float* v  = (const float*)d_in[2];
  const float* Wq = (const float*)d_in[3];
  const float* bq = (const float*)d_in[4];
  const float* Wk = (const float*)d_in[5];
  const float* bk = (const float*)d_in[6];
  const float* Wv = (const float*)d_in[7];
  const float* bv = (const float*)d_in[8];
  const float* Wo = (const float*)d_in[9];
  const float* bo = (const float*)d_in[10];
  float* out = (float*)d_out;

  const size_t NE = (size_t)Bn * Tn * Dn;        // 8M elements, 16MB as f16
  _Float16* Qh = (_Float16*)d_ws;                // [B,H,T,HD]
  _Float16* Kh = Qh + NE;                        // [B,H,T,HD]
  _Float16* Vt = Kh + NE;                        // [B,H,HD,T]
  _Float16* AO = Vt + NE;                        // [B,T,D] merged heads
  // total workspace use: 64MB

  dim3 blk(256);
  dim3 grd(64, 16);   // proj: 8192/(16*8) x 1024/64 ; attn: 64 bh x 2048/(16*8)

  proj_kernel<0><<<grd, blk, 0, stream>>>(q, Wq, bq, Qh);
  proj_kernel<1><<<grd, blk, 0, stream>>>(k, Wk, bk, Kh);
  proj_kernel<2><<<grd, blk, 0, stream>>>(v, Wv, bv, Vt);
  attn_kernel<<<grd, blk, 0, stream>>>(Qh, Kh, Vt, AO);
  oproj_kernel<<<grd, blk, 0, stream>>>(AO, Wo, bo, out);
}
